// EdgePredLayer_37812892074615
// MI455X (gfx1250) — matrix operand order repaired and checked by reading, not compiled, not run on hardware
//
#include <hip/hip_runtime.h>
#include <math.h>

// ---------- types ----------
typedef __attribute__((ext_vector_type(16))) __bf16 v16bf;
typedef __attribute__((ext_vector_type(8)))  float  v8f;
typedef unsigned int v4u __attribute__((ext_vector_type(4)));
typedef unsigned int u32x4 __attribute__((ext_vector_type(4)));
typedef int i32x4 __attribute__((ext_vector_type(4)));
typedef int i32x8 __attribute__((ext_vector_type(8)));

union FragBF16 { v16bf v; v4u q[2]; };

__device__ __forceinline__ unsigned short f32_to_bf16(float f) {
  union { float f; unsigned u; } v; v.f = f;
  unsigned r = v.u + 0x7FFFu + ((v.u >> 16) & 1u);
  return (unsigned short)(r >> 16);
}
__device__ __forceinline__ float bf16_to_f32(unsigned short h) {
  union { unsigned u; float f; } v; v.u = ((unsigned)h) << 16;
  return v.f;
}

// A fragment: 16x32 bf16 (ISA 7.12.2 16-bit A layout).
__device__ __forceinline__ v16bf load_a(const unsigned short* base, int ld,
                                        int row, int k0, int lane) {
  const unsigned short* p = base + (long)row * ld + k0 + ((lane & 16) ? 8 : 0);
  FragBF16 f;
  f.q[0] = *(const v4u*)p;
  f.q[1] = *(const v4u*)(p + 16);
  return f.v;
}
// B fragment: 32x16 bf16 K x N stored as [n][k].
__device__ __forceinline__ v16bf load_b(const unsigned short* base, int ld,
                                        int col, int k0, int lane) {
  const unsigned short* p = base + (long)col * ld + k0 + ((lane & 16) ? 8 : 0);
  FragBF16 f;
  f.q[0] = *(const v4u*)p;
  f.q[1] = *(const v4u*)(p + 16);
  return f.v;
}

// ---------- Tensor Data Mover staging (guarded), fallback = coop copy ----------
#if defined(__has_builtin)
#if __has_builtin(__builtin_amdgcn_tensor_load_to_lds) && \
    __has_builtin(__builtin_amdgcn_s_wait_tensorcnt)
#define HAVE_TDM 1
#endif
#endif

#ifdef HAVE_TDM
typedef const __attribute__((address_space(3))) void* lds_cvptr;
__device__ __forceinline__ unsigned lds_addr_of(const void* p) {
  return (unsigned)(unsigned long long)(lds_cvptr)p;
}
// 2D bf16 tile load (tile row length == tensor row length), packed into LDS.
// D# packing per CDNA5 ISA 8.3/8.4.
__device__ __forceinline__ void tdm_load_2d_bf16(const void* gsrc, unsigned ldsAddr,
                                                 unsigned rowElems, unsigned rows,
                                                 unsigned strideElems,
                                                 unsigned tensorRows) {
  unsigned long long ga = (unsigned long long)gsrc;
  u32x4 g0;
  g0.x = 1u;                                        // count=1 (valid user D#)
  g0.y = ldsAddr;                                   // lds_addr [63:32]
  g0.z = (unsigned)(ga & 0xFFFFFFFFull);            // global_addr [95:64]
  g0.w = (unsigned)((ga >> 32) & 0x01FFFFFFull)     // global_addr [120:96]
         | (2u << 30);                              // type=2 ("image")
  i32x8 g1;
  g1[0] = (int)(1u << 16);                          // data_size=1 -> 2 bytes
  g1[1] = (int)((rowElems & 0xFFFFu) << 16);        // tensor_dim0 lo16
  g1[2] = (int)(((rowElems >> 16) & 0xFFFFu) |
                ((tensorRows & 0xFFFFu) << 16));    // dim0 hi16 | dim1 lo16
  g1[3] = (int)(((tensorRows >> 16) & 0xFFFFu) |
                ((rowElems & 0xFFFFu) << 16));      // dim1 hi16 | tile_dim0
  g1[4] = (int)(rows & 0xFFFFu);                    // tile_dim1 (tile_dim2=0)
  g1[5] = (int)strideElems;                         // tensor_dim0_stride lo32
  g1[6] = 0;
  g1[7] = 0;
  i32x4 z4 = {0, 0, 0, 0};
#if __clang_major__ >= 23
  i32x8 z8 = {0, 0, 0, 0, 0, 0, 0, 0};
  __builtin_amdgcn_tensor_load_to_lds(g0, g1, z4, z4, z8, 0);
#else
  __builtin_amdgcn_tensor_load_to_lds(g0, g1, z4, z4, 0);
#endif
}
#endif  // HAVE_TDM

__device__ __forceinline__ void stage_tile(const unsigned short* g,
                                           unsigned short* lds, unsigned rowElems,
                                           unsigned rows, unsigned strideElems,
                                           unsigned tensorRows) {
#ifdef HAVE_TDM
  if (threadIdx.x < 32) {                 // wave 0 issues the TDM op
    tdm_load_2d_bf16(g, lds_addr_of(lds), rowElems, rows, strideElems, tensorRows);
    __builtin_amdgcn_s_wait_tensorcnt(0);
  }
  __syncthreads();
#else
  const v4u* s = (const v4u*)g;
  v4u* d = (v4u*)lds;
  int n = (int)(rows * rowElems) >> 3;
  for (int i = threadIdx.x; i < n; i += blockDim.x) d[i] = s[i];
  __syncthreads();
#endif
}

// ---------- kernel 1: fp32 -> bf16 cast ----------
__global__ void cast_f32_bf16(const float* __restrict__ in,
                              unsigned short* __restrict__ out, int n) {
  int i = blockIdx.x * blockDim.x + threadIdx.x;
  int stride = gridDim.x * blockDim.x;
  for (; i < n; i += stride) out[i] = f32_to_bf16(in[i]);
}

// ---------- kernel 2: tiled transpose + cast ----------
__global__ void transpose_cast(const float* __restrict__ src,
                               unsigned short* __restrict__ dst, int R, int C) {
  __shared__ float t[32][33];
  long b = blockIdx.z;
  const float* s = src + b * (long)R * C;
  unsigned short* d = dst + b * (long)R * C;
  int r0 = blockIdx.y * 32, c0 = blockIdx.x * 32;
  int tx = threadIdx.x, ty = threadIdx.y;  // block (32, 8)
  for (int i = 0; i < 32; i += 8) {
    int r = r0 + ty + i, c = c0 + tx;
    if (r < R && c < C) t[ty + i][tx] = s[(long)r * C + c];
  }
  __syncthreads();
  for (int i = 0; i < 32; i += 8) {
    int c = c0 + ty + i, r = r0 + tx;
    if (c < C && r < R) d[(long)c * R + r] = f32_to_bf16(t[tx][ty + i]);
  }
}

// ---------- kernel 3: WMMA GEMM + bias; WT tile staged in LDS ----------
// O[M x N] = X[M x Kd] @ W + bias, W transposed as WT[N x Kd].  Kd <= 256.
__global__ __launch_bounds__(256) void gemm_bias_bf16(
    const unsigned short* __restrict__ X, const unsigned short* __restrict__ WT,
    const float* __restrict__ bias, unsigned short* __restrict__ O,
    unsigned short* __restrict__ OT, int M, int N, int Kd, float scale) {
  __shared__ unsigned short ws[128 * 256];  // 64 KB: 128 cols x Kd
  int lane = threadIdx.x & 31, wave = threadIdx.x >> 5;
  int r0 = blockIdx.x * 128 + wave * 16;
  int c0 = blockIdx.y * 128;
  int rowA = r0 + (lane & 15);
  stage_tile(WT + (long)c0 * Kd, ws, Kd, 128, Kd, N);

  v8f zero = {};
  v8f acc[8];
#pragma unroll
  for (int t = 0; t < 8; ++t) acc[t] = zero;

  // main loop with unconditional A-prefetch; last k-step peeled
  v16bf a0 = load_a(X, Kd, rowA, 0, lane);
  int k = 0;
  for (; k + 32 < Kd; k += 32) {
    v16bf a1 = load_a(X, Kd, rowA, k + 32, lane);
    v16bf bt[8];
#pragma unroll
    for (int t = 0; t < 8; ++t)
      bt[t] = load_b(ws, Kd, t * 16 + (lane & 15), k, lane);
#pragma unroll
    for (int t = 0; t < 8; ++t)
      acc[t] = __builtin_amdgcn_wmma_f32_16x16x32_bf16(
          false, a0, false, bt[t], (short)0, acc[t], false, false);
    a0 = a1;
  }
  {
    v16bf bt[8];
#pragma unroll
    for (int t = 0; t < 8; ++t)
      bt[t] = load_b(ws, Kd, t * 16 + (lane & 15), k, lane);
#pragma unroll
    for (int t = 0; t < 8; ++t)
      acc[t] = __builtin_amdgcn_wmma_f32_16x16x32_bf16(
          false, a0, false, bt[t], (short)0, acc[t], false, false);
  }
  int rb = (lane & 16) ? 8 : 0;
#pragma unroll
  for (int t = 0; t < 8; ++t) {
    int col = c0 + t * 16 + (lane & 15);
    float bv = bias ? bias[col] : 0.f;
#pragma unroll
    for (int i = 0; i < 8; ++i) {
      int row = r0 + i + rb;
      unsigned short h = f32_to_bf16((acc[t][i] + bv) * scale);
      if (O)  O[(long)row * N + col] = h;
      if (OT) OT[(long)col * M + row] = h;
    }
  }
}

// ---------- kernel 4: flash attention; Q tile staged in LDS ----------
__global__ __launch_bounds__(256) void flash_attn(
    const unsigned short* __restrict__ Q, const unsigned short* __restrict__ Km,
    const unsigned short* __restrict__ VT, unsigned short* __restrict__ AE,
    int Nq, int Nk, int Af) {
  __shared__ unsigned short qs[16 * 1024];  // 32 KB
  __shared__ float sc[16][128];
  __shared__ unsigned short pm[16][128];
  __shared__ float mrow[16], lrow[16], crow[16];
  int lane = threadIdx.x & 31, wave = threadIdx.x >> 5;
  int q0 = blockIdx.x * 16;
  if (threadIdx.x < 16) { mrow[threadIdx.x] = -3.0e38f; lrow[threadIdx.x] = 0.f; }
  stage_tile(Q + (long)q0 * Af, qs, Af, 16, Af, Nq);

  v8f zero = {};
  v8f acc[8];
#pragma unroll
  for (int t = 0; t < 8; ++t) acc[t] = zero;
  int rb = (lane & 16) ? 8 : 0;

  for (int j0 = 0; j0 < Nk; j0 += 128) {
    // ---- S tile (16 q-rows x 16 keys per wave), 1-deep pipelined
    v8f s = zero;
    int key = j0 + wave * 16 + (lane & 15);
    v16bf a0 = load_a(qs, Af, lane & 15, 0, lane);
    v16bf b0 = load_b(Km, Af, key, 0, lane);
    for (int k = 32; k < Af; k += 32) {
      v16bf a1 = load_a(qs, Af, lane & 15, k, lane);
      v16bf b1 = load_b(Km, Af, key, k, lane);
      s = __builtin_amdgcn_wmma_f32_16x16x32_bf16(false, a0, false, b0,
                                                  (short)0, s, false, false);
      a0 = a1; b0 = b1;
    }
    s = __builtin_amdgcn_wmma_f32_16x16x32_bf16(false, a0, false, b0, (short)0,
                                                s, false, false);
#pragma unroll
    for (int i = 0; i < 8; ++i)
      sc[i + rb][wave * 16 + (lane & 15)] = s[i];
    __syncthreads();

    // ---- online softmax over this 128-key slab (wave 0, 1 lane per row)
    if (wave == 0 && lane < 16) {
      int r = lane;
      float m0 = mrow[r], tm = m0;
      for (int j = 0; j < 128; ++j) tm = fmaxf(tm, sc[r][j]);
      float cr = __expf(m0 - tm);
      float l = lrow[r] * cr;
      for (int j = 0; j < 128; ++j) {
        float e = __expf(sc[r][j] - tm);
        l += e;
        pm[r][j] = f32_to_bf16(e);
      }
      mrow[r] = tm; lrow[r] = l; crow[r] = cr;
    }
    __syncthreads();

    // ---- rescale accumulators
    {
      float cr[8];
#pragma unroll
      for (int i = 0; i < 8; ++i) cr[i] = crow[i + rb];
#pragma unroll
      for (int t = 0; t < 8; ++t)
#pragma unroll
        for (int i = 0; i < 8; ++i) acc[t][i] *= cr[i];
    }
    // ---- O += P @ V (this wave owns f-cols [128*wave, 128*wave+128))
    for (int ks = 0; ks < 128; ks += 32) {
      v16bf a = load_a(&pm[0][0], 128, lane & 15, ks, lane);
      v16bf bt[8];
#pragma unroll
      for (int t = 0; t < 8; ++t)
        bt[t] = load_b(VT, Nk, wave * 128 + t * 16 + (lane & 15), j0 + ks, lane);
#pragma unroll
      for (int t = 0; t < 8; ++t)
        acc[t] = __builtin_amdgcn_wmma_f32_16x16x32_bf16(
            false, a, false, bt[t], (short)0, acc[t], false, false);
    }
    __syncthreads();
  }
#pragma unroll
  for (int t = 0; t < 8; ++t) {
    int f = wave * 128 + t * 16 + (lane & 15);
#pragma unroll
    for (int i = 0; i < 8; ++i) {
      int r = i + rb;
      AE[(long)(q0 + r) * Af + f] = f32_to_bf16(acc[t][i] / lrow[r]);
    }
  }
}

// ---------- kernel 5: gather edge endpoints ----------
__global__ void gather_edges(const long long* __restrict__ data,
                             const unsigned short* __restrict__ AE,
                             unsigned short* __restrict__ Hu,
                             unsigned short* __restrict__ Hv, int Af) {
  int e = blockIdx.x;
  long u = (long)data[(long)e * 2 + 0];
  long v = (long)data[(long)e * 2 + 1];
  for (int c = threadIdx.x; c < Af; c += blockDim.x) {
    Hu[(long)e * Af + c] = AE[u * Af + c];
    Hv[(long)e * Af + c] = AE[v * Af + c];
  }
}

// ---------- kernel 6: bilinear edge scores; Hu staged in LDS ----------
// out[e,k] = h_u[e]^T W_k h_v[e].  Each wave owns TWO 16-col f-tiles
// (f and f+128): two independent WMMA chains, shared A fragment, 1-deep
// global B pipeline.
__global__ __launch_bounds__(256) void edge_bilinear(
    const unsigned short* __restrict__ Hu, const unsigned short* __restrict__ Hv,
    const unsigned short* __restrict__ WeT, float* __restrict__ out, int E,
    int KK, int Af) {
  __shared__ unsigned short hs[16 * 1024];  // 32 KB
  __shared__ float rsum[16];
  int lane = threadIdx.x & 31, wave = threadIdx.x >> 5;
  int e0 = blockIdx.x * 16;
  int k = blockIdx.y;
  const unsigned short* Wk = WeT + (long)k * Af * Af;
  if (threadIdx.x < 16) rsum[threadIdx.x] = 0.f;
  stage_tile(Hu + (long)e0 * Af, hs, Af, 16, Af, E);

  int rb = (lane & 16) ? 8 : 0;
  float part[8];
#pragma unroll
  for (int i = 0; i < 8; ++i) part[i] = 0.f;
  v8f zero = {};
  for (int fc = 0; fc < Af; fc += 256) {
    int f0 = fc + wave * 16 + (lane & 15);
    int f1 = f0 + 128;
    __builtin_prefetch(Wk + (long)(f0 + 256) * Af, 0, 0);  // next fc pass
    v8f acc0 = zero, acc1 = zero;
    v16bf a0  = load_a(hs, Af, lane & 15, 0, lane);
    v16bf b00 = load_b(Wk, Af, f0, 0, lane);
    v16bf b10 = load_b(Wk, Af, f1, 0, lane);
    int d = 0;
    for (; d + 32 < Af; d += 32) {
      v16bf a1  = load_a(hs, Af, lane & 15, d + 32, lane);
      v16bf b01 = load_b(Wk, Af, f0, d + 32, lane);
      v16bf b11 = load_b(Wk, Af, f1, d + 32, lane);
      acc0 = __builtin_amdgcn_wmma_f32_16x16x32_bf16(false, a0, false, b00,
                                                     (short)0, acc0, false, false);
      acc1 = __builtin_amdgcn_wmma_f32_16x16x32_bf16(false, a0, false, b10,
                                                     (short)0, acc1, false, false);
      a0 = a1; b00 = b01; b10 = b11;
    }
    acc0 = __builtin_amdgcn_wmma_f32_16x16x32_bf16(false, a0, false, b00,
                                                   (short)0, acc0, false, false);
    acc1 = __builtin_amdgcn_wmma_f32_16x16x32_bf16(false, a0, false, b10,
                                                   (short)0, acc1, false, false);
#pragma unroll
    for (int i = 0; i < 8; ++i) {
      int r = i + rb;
      float hv0 = bf16_to_f32(Hv[(long)(e0 + r) * Af + f0]);
      float hv1 = bf16_to_f32(Hv[(long)(e0 + r) * Af + f1]);
      part[i] += acc0[i] * hv0 + acc1[i] * hv1;
    }
  }
#pragma unroll
  for (int i = 0; i < 8; ++i) atomicAdd(&rsum[i + rb], part[i]);
  __syncthreads();
  if (threadIdx.x < 16) out[(long)(e0 + threadIdx.x) * KK + k] = rsum[threadIdx.x];
}

// ---------- host launcher ----------
extern "C" void kernel_launch(void* const* d_in, const int* in_sizes, int n_in,
                              void* d_out, int out_size, void* d_ws, size_t ws_size,
                              hipStream_t stream) {
  const int N = 8192, D = 256, A = 1024, KK = 65, E = 4096;
  const float* emb = (const float*)d_in[0];
  const long long* data = (const long long*)d_in[1];
  const float* Wq = (const float*)d_in[2];
  const float* bq = (const float*)d_in[3];
  const float* Wk = (const float*)d_in[4];
  const float* bk = (const float*)d_in[5];
  const float* Wv = (const float*)d_in[6];
  const float* bv = (const float*)d_in[7];
  const float* We = (const float*)d_in[8];
  float* out = (float*)d_out;

  char* ws = (char*)d_ws;
  size_t off = 0;
  auto carve = [&](size_t bytes) {
    char* p = ws + off;
    off += (bytes + 255) & ~(size_t)255;
    return p;
  };
  unsigned short* embB = (unsigned short*)carve((size_t)N * D * 2);
  unsigned short* WqT  = (unsigned short*)carve((size_t)D * A * 2);
  unsigned short* WkT  = (unsigned short*)carve((size_t)D * A * 2);
  unsigned short* WvT  = (unsigned short*)carve((size_t)D * A * 2);
  unsigned short* Qb   = (unsigned short*)carve((size_t)N * A * 2);
  unsigned short* Kb   = (unsigned short*)carve((size_t)N * A * 2);
  unsigned short* VTb  = (unsigned short*)carve((size_t)A * N * 2);
  unsigned short* AEb  = (unsigned short*)carve((size_t)N * A * 2);
  unsigned short* Hub  = (unsigned short*)carve((size_t)E * A * 2);
  unsigned short* Hvb  = (unsigned short*)carve((size_t)E * A * 2);
  unsigned short* WeT  = (unsigned short*)carve((size_t)KK * A * A * 2);
  (void)ws_size; (void)in_sizes; (void)n_in; (void)out_size;

  cast_f32_bf16<<<2048, 256, 0, stream>>>(emb, embB, N * D);

  transpose_cast<<<dim3(A / 32, D / 32, 1), dim3(32, 8), 0, stream>>>(Wq, WqT, D, A);
  transpose_cast<<<dim3(A / 32, D / 32, 1), dim3(32, 8), 0, stream>>>(Wk, WkT, D, A);
  transpose_cast<<<dim3(A / 32, D / 32, 1), dim3(32, 8), 0, stream>>>(Wv, WvT, D, A);
  transpose_cast<<<dim3(A / 32, A / 32, KK), dim3(32, 8), 0, stream>>>(We, WeT, A, A);

  dim3 ggrid(N / 128, A / 128);
  gemm_bias_bf16<<<ggrid, 256, 0, stream>>>(embB, WqT, bq, Qb, nullptr, N, A, D, 0.0625f);
  gemm_bias_bf16<<<ggrid, 256, 0, stream>>>(embB, WkT, bk, Kb, nullptr, N, A, D, 1.0f);
  gemm_bias_bf16<<<ggrid, 256, 0, stream>>>(embB, WvT, bv, nullptr, VTb, N, A, D, 1.0f);

  flash_attn<<<N / 16, 256, 0, stream>>>(Qb, Kb, VTb, AEb, N, N, A);

  gather_edges<<<E, 256, 0, stream>>>(data, AEb, Hub, Hvb, A);

  edge_bilinear<<<dim3(E / 16, KK), 256, 0, stream>>>(Hub, Hvb, WeT, out, E, KK, A);
}
